// Model_62113817035535
// MI455X (gfx1250) — compile-verified, matrix operand order checked by reading, not run on hardware
//
#include <hip/hip_runtime.h>
#include <hip/hip_bf16.h>
#include <math.h>

// ---------------------------------------------------------------------------
// MixSTE/HoT-style pose transformer forward on gfx1250 (MI455X).
// All big GEMMs: v_wmma_f32_16x16x32_bf16, fp32 accum.  Weights are converted
// to a bf16 arena ONCE per launch; producers (LN / attention / GEMM epilogue)
// emit bf16 wherever the consumer is a GEMM, so the WMMA k-loop is pure data
// movement.  A-tiles are staged with CDNA5 async global->LDS DMA (ASYNCcnt),
// B-tiles with a transpose scatter.  Streams/residuals/LN stats stay fp32.
// ---------------------------------------------------------------------------

typedef __attribute__((ext_vector_type(16))) __bf16 v16bf;
typedef __attribute__((ext_vector_type(8)))  __bf16 v8bf;
typedef __attribute__((ext_vector_type(8)))  float  v8f;

#define C_DIM   512
#define NH_DIM  8
#define HD_DIM  64
#define HID_DIM 1024
#define C3_DIM  1536
#define DEPTH_N 8
#define B_DIM   8
#define F0_DIM  243
#define N_DIM   17
#define TOKK    81
#define CENTER_F 121

__device__ __forceinline__ __bf16 f2bf(float f) {
    unsigned u = __builtin_bit_cast(unsigned, f);
    u += 0x7FFFu + ((u >> 16) & 1u);      // round-to-nearest-even
    unsigned short h = (unsigned short)(u >> 16);
    return __builtin_bit_cast(__bf16, h);
}

__device__ __forceinline__ float bf2f(__bf16 h) {
    unsigned short b = __builtin_bit_cast(unsigned short, h);
    unsigned u = ((unsigned)b) << 16;
    return __builtin_bit_cast(float, u);
}

__device__ __forceinline__ float gelu_exact(float x) {
    return 0.5f * x * (1.0f + erff(x * 0.70710678118654752440f));
}

// ---------------------------------------------------------------------------
// fp32 -> bf16 bulk convert (n must be a multiple of 8)
// ---------------------------------------------------------------------------
__global__ __launch_bounds__(256)
void cvt_kernel(const float* __restrict__ in, __bf16* __restrict__ out, size_t n8)
{
    size_t t = (size_t)blockIdx.x * 256 + threadIdx.x;
    if (t >= n8) return;
    const float4* src = (const float4*)(in + t * 8);
    float4 a = src[0], b = src[1];
    v8bf p;
    p[0] = f2bf(a.x); p[1] = f2bf(a.y); p[2] = f2bf(a.z); p[3] = f2bf(a.w);
    p[4] = f2bf(b.x); p[5] = f2bf(b.y); p[6] = f2bf(b.z); p[7] = f2bf(b.w);
    *(v8bf*)(out + t * 8) = p;
}

// ---------------------------------------------------------------------------
// WMMA GEMM: out = act(X[M,K] @ W[K,N] + bias) (+ residual)
// X, W bf16; Y (fp32) and/or Ybf (bf16) outputs selected by non-null pointer.
// REQUIRES: K % 32 == 0, N % 128 == 0 (true for all call sites here).
// BM=BN=128, BK=32.  256 threads = 8 waves; each wave owns a 32x64 out tile
// (2x4 of 16x16 WMMA frags) -> 64 WMMAs per 16KB of staged LDS.
// A staged [128][32] row-major via GLOBAL_LOAD_ASYNC_TO_LDS_B128 (per-lane
// 16B address-to-address DMA, ISA 15.18.3; tracked by ASYNCcnt); B staged
// TRANSPOSED [128(N)][32(K)] so fragment loads are contiguous ds_load_b128s.
// Fragment layout per CDNA5 ISA (16-bit A 16x32, wave32):
//   lane L: row M = L%16;  K = khalf..khalf+7 and 16+khalf..16+khalf+7,
//   khalf = (L<16) ? 0 : 8.  B mirrors with N = L%16.
// ---------------------------------------------------------------------------
__global__ __launch_bounds__(256)
void wmma_gemm_kernel(const __bf16* __restrict__ X, const __bf16* __restrict__ W,
                      const float* __restrict__ bias, const float* __restrict__ residual,
                      float* __restrict__ Y, __bf16* __restrict__ Ybf,
                      int M, int N, int K, int act)
{
    __shared__ __bf16 As[128 * 32];
    __shared__ __bf16 Bs[128 * 32];   // [n][k]

    const int bm   = blockIdx.y * 128;
    const int bn   = blockIdx.x * 128;
    const int tid  = threadIdx.x;
    const int wave = tid >> 5;
    const int lane = tid & 31;
    const int wm   = (wave >> 1) * 32;    // 0..96
    const int wn   = (wave & 1) * 64;     // 0 or 64
    const int m16  = lane & 15;
    const int khalf = (lane >> 4) << 3;   // 0 or 8

    // A-staging coords: each thread covers 16 consecutive K of one row
    const int ar = tid >> 1;              // 0..127
    const int ac = (tid & 1) * 16;        // 0 or 16
    int agm = bm + ar; if (agm >= M) agm = M - 1;   // clamp; dup rows discarded

    // Wave-relative LDS byte offset = low 32 bits of the generic LDS address
    // (ISA 10.2: LDS aperture addresses truncate to the LDS offset).
    const unsigned a_lds = (unsigned)(size_t)&As[ar * 32 + ac];

    v8f acc[2][4] = {};

    for (int k0 = 0; k0 < K; k0 += 32) {
        // ---- stage A: async global->LDS DMA, 2x b128 per lane ----
        {
            const void* gsrc = (const void*)(X + (size_t)agm * K + k0 + ac);
            asm volatile(
                "global_load_async_to_lds_b128 %0, %1, off\n\t"
                "global_load_async_to_lds_b128 %0, %1, off offset:16"
                :: "v"(a_lds), "v"(gsrc) : "memory");
        }
        // ---- stage B transposed: coalesced b128 reads, b16 scatter ----
#pragma unroll
        for (int it = 0; it < 2; ++it) {
            int slot = tid + it * 256;        // 0..511
            int kk = slot >> 4;               // 0..31
            int nc = (slot & 15) * 8;         // 0..120
            v8bf w = *(const v8bf*)(W + (size_t)(k0 + kk) * N + bn + nc);
            __bf16* dst = &Bs[nc * 32 + kk];  // Bs[n][k], stride 32
            dst[0 * 32] = w[0]; dst[1 * 32] = w[1]; dst[2 * 32] = w[2]; dst[3 * 32] = w[3];
            dst[4 * 32] = w[4]; dst[5 * 32] = w[5]; dst[6 * 32] = w[6]; dst[7 * 32] = w[7];
        }
        // prefetch next K tile (-> global_prefetch_b8)
        if (k0 + 32 < K) {
            __builtin_prefetch(X + (size_t)agm * K + k0 + 32 + ac, 0, 0);
            __builtin_prefetch(W + (size_t)(k0 + 32 + (tid >> 3)) * N + bn, 0, 0);
        }
        // async copies are tracked by ASYNCcnt, not covered by the barrier's
        // dscnt wait -> drain before publishing the tile
        asm volatile("s_wait_asynccnt 0x0" ::: "memory");
        __syncthreads();

        v16bf a[2], b[4];
#pragma unroll
        for (int mi = 0; mi < 2; ++mi) {
            const __bf16* base = &As[(wm + mi * 16 + m16) * 32];
            v8bf lo = *(const v8bf*)(base + khalf);
            v8bf hi = *(const v8bf*)(base + 16 + khalf);
            a[mi] = __builtin_shufflevector(lo, hi, 0,1,2,3,4,5,6,7,8,9,10,11,12,13,14,15);
        }
#pragma unroll
        for (int ni = 0; ni < 4; ++ni) {
            const __bf16* base = &Bs[(wn + ni * 16 + m16) * 32];
            v8bf lo = *(const v8bf*)(base + khalf);
            v8bf hi = *(const v8bf*)(base + 16 + khalf);
            b[ni] = __builtin_shufflevector(lo, hi, 0,1,2,3,4,5,6,7,8,9,10,11,12,13,14,15);
        }
#pragma unroll
        for (int mi = 0; mi < 2; ++mi)
#pragma unroll
            for (int ni = 0; ni < 4; ++ni)
                acc[mi][ni] = __builtin_amdgcn_wmma_f32_16x16x32_bf16(
                    false, a[mi], false, b[ni], (short)0, acc[mi][ni], false, false);
        __syncthreads();
    }

    // ---- epilogue: C/D layout: VGPR r -> M = r + 8*(lane>=16); N = lane%16 ----
#pragma unroll
    for (int mi = 0; mi < 2; ++mi)
#pragma unroll
        for (int ni = 0; ni < 4; ++ni) {
            int col = bn + wn + ni * 16 + m16;           // always < N (N%128==0)
            float bv = bias ? bias[col] : 0.0f;
#pragma unroll
            for (int r = 0; r < 8; ++r) {
                int row = bm + wm + mi * 16 + r + khalf;
                if (row < M) {
                    float v = acc[mi][ni][r] + bv;
                    if (act == 1) v = gelu_exact(v);
                    if (residual) v += residual[(size_t)row * N + col];
                    if (Y)   Y[(size_t)row * N + col] = v;
                    if (Ybf) Ybf[(size_t)row * N + col] = f2bf(v);
                }
            }
        }
}

// ---------------------------------------------------------------------------
// Embedding: h[t,j,c] = x0*W[0,c] + x1*W[1,c] + eb[c] + spos[j,c]
// ---------------------------------------------------------------------------
__global__ void embed_kernel(const float* __restrict__ x, const float* __restrict__ ew,
                             const float* __restrict__ eb, const float* __restrict__ spos,
                             float* __restrict__ h, int tokens)
{
    size_t tid = (size_t)blockIdx.x * blockDim.x + threadIdx.x;
    size_t tot = (size_t)tokens * C_DIM;
    if (tid >= tot) return;
    int c = (int)(tid % C_DIM);
    size_t tok = tid / C_DIM;
    int j = (int)(tok % N_DIM);
    float x0 = x[tok * 2 + 0], x1 = x[tok * 2 + 1];
    h[tid] = x0 * ew[c] + x1 * ew[C_DIM + c] + eb[c] + spos[(size_t)j * C_DIM + c];
}

// ---------------------------------------------------------------------------
// LayerNorm over last dim (C=512): one block (256 thr) per row.
// f32-out version (stream norms, may alias) and bf16-out version (GEMM feeds).
// ---------------------------------------------------------------------------
__device__ __forceinline__ void ln_stats(const float* xr, float* red, float& mean, float& inv, float eps)
{
    float ls = 0.f;
    for (int c = threadIdx.x; c < C_DIM; c += 256) ls += xr[c];
    red[threadIdx.x] = ls; __syncthreads();
    for (int off = 128; off > 0; off >>= 1) {
        if ((int)threadIdx.x < off) red[threadIdx.x] += red[threadIdx.x + off];
        __syncthreads();
    }
    mean = red[0] * (1.0f / C_DIM);
    __syncthreads();
    float lv = 0.f;
    for (int c = threadIdx.x; c < C_DIM; c += 256) { float d = xr[c] - mean; lv += d * d; }
    red[threadIdx.x] = lv; __syncthreads();
    for (int off = 128; off > 0; off >>= 1) {
        if ((int)threadIdx.x < off) red[threadIdx.x] += red[threadIdx.x + off];
        __syncthreads();
    }
    inv = rsqrtf(red[0] * (1.0f / C_DIM) + eps);
}

__global__ __launch_bounds__(256)
void ln_kernel(const float* __restrict__ x, const float* __restrict__ s,
               const float* __restrict__ b, float* __restrict__ y, float eps)
{
    __shared__ float red[256];
    const size_t row = blockIdx.x;
    const float* xr = x + row * C_DIM;
    float mean, inv;
    ln_stats(xr, red, mean, inv, eps);
    for (int c = threadIdx.x; c < C_DIM; c += 256)
        y[row * C_DIM + c] = (xr[c] - mean) * inv * s[c] + b[c];
}

__global__ __launch_bounds__(256)
void ln_bf16_kernel(const float* __restrict__ x, const float* __restrict__ s,
                    const float* __restrict__ b, __bf16* __restrict__ y, float eps)
{
    __shared__ float red[256];
    const size_t row = blockIdx.x;
    const float* xr = x + row * C_DIM;
    float mean, inv;
    ln_stats(xr, red, mean, inv, eps);
    for (int c = threadIdx.x; c < C_DIM; c += 256)
        y[row * C_DIM + c] = f2bf((xr[c] - mean) * inv * s[c] + b[c]);
}

// ---------------------------------------------------------------------------
// Attention from packed bf16 qkv [SB*N, 3C]: q col = h*64+d, k = 512+...,
// v = 1024+...  One block per (seq, head); each thread owns queries
// (2-pass online softmax, fp32 math).  Output bf16 (feeds proj GEMM).
// ---------------------------------------------------------------------------
__global__ __launch_bounds__(64)
void attention_kernel(const __bf16* __restrict__ qkv, __bf16* __restrict__ out,
                      int Ntok)
{
    const int seq = blockIdx.x;
    const int h   = blockIdx.y;
    const __bf16* base = qkv + (size_t)seq * Ntok * C3_DIM;
    const float scale = 0.125f;   // hd=64 ** -0.5

    for (int i = threadIdx.x; i < Ntok; i += 64) {
        const __bf16* qi = base + (size_t)i * C3_DIM + h * HD_DIM;
        float q[HD_DIM];
#pragma unroll
        for (int d = 0; d < HD_DIM; ++d) q[d] = bf2f(qi[d]);

        float mx = -3.4e38f;
        for (int j = 0; j < Ntok; ++j) {
            const __bf16* kj = base + (size_t)j * C3_DIM + C_DIM + h * HD_DIM;
            float s = 0.f;
#pragma unroll
            for (int d = 0; d < HD_DIM; ++d) s += q[d] * bf2f(kj[d]);
            mx = fmaxf(mx, s * scale);
        }
        float o[HD_DIM];
#pragma unroll
        for (int d = 0; d < HD_DIM; ++d) o[d] = 0.f;
        float denom = 0.f;
        for (int j = 0; j < Ntok; ++j) {
            const __bf16* kj = base + (size_t)j * C3_DIM + C_DIM + h * HD_DIM;
            const __bf16* vj = base + (size_t)j * C3_DIM + 2 * C_DIM + h * HD_DIM;
            float s = 0.f;
#pragma unroll
            for (int d = 0; d < HD_DIM; ++d) s += q[d] * bf2f(kj[d]);
            float e = __expf(s * scale - mx);
            denom += e;
#pragma unroll
            for (int d = 0; d < HD_DIM; ++d) o[d] += e * bf2f(vj[d]);
        }
        float invd = 1.0f / denom;
        __bf16* op = out + ((size_t)seq * Ntok + i) * C_DIM + h * HD_DIM;
#pragma unroll
        for (int d = 0; d < HD_DIM; ++d) op[d] = f2bf(o[d] * invd);
    }
}

// ---------------------------------------------------------------------------
// [B, D1, D2, C] -> [B, D2, D1, C]
// ---------------------------------------------------------------------------
__global__ void permute_kernel(const float* __restrict__ in, float* __restrict__ out,
                               int B, int D1, int D2)
{
    size_t tid = (size_t)blockIdx.x * blockDim.x + threadIdx.x;
    size_t tot = (size_t)B * D1 * D2 * C_DIM;
    if (tid >= tot) return;
    int c = (int)(tid % C_DIM);
    size_t r = tid / C_DIM;
    int d2 = (int)(r % D2); r /= D2;
    int d1 = (int)(r % D1);
    int b  = (int)(r / D1);
    out[(((size_t)b * D2 + d2) * D1 + d1) * C_DIM + c] = in[tid];
}

__global__ void tpos_add_kernel(float* __restrict__ h, const float* __restrict__ tpos,
                                int S, int F)
{
    size_t tid = (size_t)blockIdx.x * blockDim.x + threadIdx.x;
    size_t tot = (size_t)S * F * C_DIM;
    if (tid >= tot) return;
    int c = (int)(tid % C_DIM);
    int t = (int)((tid / C_DIM) % F);
    h[tid] += tpos[(size_t)t * C_DIM + c];
}

// ------------------------- DPC-KNN clustering ------------------------------
__global__ __launch_bounds__(256)
void frame_mean_kernel(const float* __restrict__ h, float* __restrict__ xk)
{
    const size_t bf = blockIdx.x;   // b*F + f
    for (int c = threadIdx.x; c < C_DIM; c += 256) {
        float s = 0.f;
        for (int j = 0; j < N_DIM; ++j) s += h[(bf * N_DIM + j) * C_DIM + c];
        xk[bf * C_DIM + c] = s * (1.0f / N_DIM);
    }
}

__global__ __launch_bounds__(64)
void dist_kernel(const float* __restrict__ xk, float* __restrict__ dist, int F)
{
    const int b = blockIdx.x, i = blockIdx.y;
    const float* xi = xk + ((size_t)b * F + i) * C_DIM;
    const float invs = rsqrtf((float)C_DIM);
    for (int j = threadIdx.x; j < F; j += 64) {
        const float* xj = xk + ((size_t)b * F + j) * C_DIM;
        float s = 0.f;
        for (int c = 0; c < C_DIM; ++c) { float d = xi[c] - xj[c]; s += d * d; }
        dist[((size_t)b * F + i) * F + j] = sqrtf(fmaxf(s, 0.f)) * invs;
    }
}

__global__ __launch_bounds__(256)
void density_kernel(const float* __restrict__ dist, const float* __restrict__ noise,
                    float* __restrict__ density, int F)
{
    const int b = blockIdx.x;
    for (int i = threadIdx.x; i < F; i += 256) {
        const float* di = dist + ((size_t)b * F + i) * F;
        float mn = 3.4e38f;
        for (int j = 0; j < F; ++j) if (j != i) mn = fminf(mn, di[j]);
        // k=2 nearest distances = {0 (self), mn}; density = exp(-mean(d^2))
        density[(size_t)b * F + i] = __expf(-(mn * mn) * 0.5f) + noise[(size_t)b * F + i] * 1e-6f;
    }
}

__global__ __launch_bounds__(256)
void dmax_kernel(const float* __restrict__ dist, float* __restrict__ dmax, int F)
{
    __shared__ float red[256];
    const int b = blockIdx.x;
    const float* d = dist + (size_t)b * F * F;
    float m = 0.f;
    for (int i = threadIdx.x; i < F * F; i += 256) m = fmaxf(m, d[i]);
    red[threadIdx.x] = m; __syncthreads();
    for (int off = 128; off > 0; off >>= 1) {
        if ((int)threadIdx.x < off) red[threadIdx.x] = fmaxf(red[threadIdx.x], red[threadIdx.x + off]);
        __syncthreads();
    }
    if (threadIdx.x == 0) dmax[b] = red[0];
}

__global__ __launch_bounds__(256)
void score_kernel(const float* __restrict__ dist, const float* __restrict__ density,
                  const float* __restrict__ dmax, float* __restrict__ score, int F)
{
    const int b = blockIdx.x;
    const float dm = dmax[b];
    for (int i = threadIdx.x; i < F; i += 256) {
        float deni = density[(size_t)b * F + i];
        const float* di = dist + ((size_t)b * F + i) * F;
        float dp = dm;
        for (int j = 0; j < F; ++j)
            if (density[(size_t)b * F + j] > deni) dp = fminf(dp, di[j]);
        float s = dp * deni;
        score[(size_t)b * F + i] = (i == CENTER_F) ? -__builtin_inff() : s;
    }
}

__global__ void topk_kernel(const float* __restrict__ score, int* __restrict__ idx, int F)
{
    const int b = blockIdx.x;
    if (threadIdx.x != 0) return;
    float s[F0_DIM];
    for (int i = 0; i < F; ++i) s[i] = score[(size_t)b * F + i];
    int out[TOKK];
    for (int k = 0; k < TOKK - 1; ++k) {           // top-80, stable tie-break
        float best = -3.4e38f; int bi = 0;
        for (int i = 0; i < F; ++i) if (s[i] > best) { best = s[i]; bi = i; }
        out[k] = bi; s[bi] = -3.4e38f;
    }
    out[TOKK - 1] = CENTER_F;
    for (int k = 1; k < TOKK; ++k) {               // insertion sort ascending
        int v = out[k], j = k - 1;
        while (j >= 0 && out[j] > v) { out[j + 1] = out[j]; --j; }
        out[j + 1] = v;
    }
    for (int k = 0; k < TOKK; ++k) idx[b * TOKK + k] = out[k];
}

__global__ void gather_kernel(const float* __restrict__ in, const int* __restrict__ idx,
                              float* __restrict__ out, int Fin)
{
    size_t tid = (size_t)blockIdx.x * blockDim.x + threadIdx.x;
    const size_t NC = (size_t)N_DIM * C_DIM;
    size_t tot = (size_t)B_DIM * TOKK * NC;
    if (tid >= tot) return;
    size_t o = tid % NC;
    size_t r = tid / NC;
    int t = (int)(r % TOKK);
    int b = (int)(r / TOKK);
    int src = idx[b * TOKK + t];
    out[tid] = in[((size_t)b * Fin + src) * NC + o];
}

// ------------------------------ head ---------------------------------------
__global__ void head_kernel(const __bf16* __restrict__ x, const float* __restrict__ w,
                            const float* __restrict__ b, float* __restrict__ out, int M)
{
    int t = blockIdx.x * blockDim.x + threadIdx.x;
    if (t >= M * 3) return;
    int j = t % 3, row = t / 3;
    const __bf16* xr = x + (size_t)row * C_DIM;
    float s = b[j];
    for (int c = 0; c < C_DIM; ++c) s += bf2f(xr[c]) * w[c * 3 + j];
    out[t] = s;
}

__global__ void idx_out_kernel(const int* __restrict__ idx, float* __restrict__ out, int n)
{
    int t = blockIdx.x * blockDim.x + threadIdx.x;
    if (t < n) out[t] = (float)idx[t];
}

// ---------------------------------------------------------------------------
// Host orchestration
// ---------------------------------------------------------------------------
struct BlkP {
    const float *n1_s, *n1_b, *qkv_b, *proj_b, *n2_s, *n2_b, *fc1_b, *fc2_b;
    const __bf16 *qkv_w, *proj_w, *fc1_w, *fc2_w;   // bf16 arena
};

static void run_block(hipStream_t st, float* x, int M, int SB, int Ntok,
                      const BlkP& P, int layer,
                      __bf16* bufT, __bf16* bufQKV, __bf16* bufAttn, __bf16* bufH1)
{
    const size_t lc = (size_t)layer * C_DIM;
    const unsigned my = (unsigned)((M + 127) / 128);
    // LN1 -> bf16
    ln_bf16_kernel<<<M, 256, 0, st>>>(x, P.n1_s + lc, P.n1_b + lc, bufT, 1e-6f);
    // qkv = LN1(x) @ qkv_w + qkv_b   (bf16 out -> attention)
    wmma_gemm_kernel<<<dim3(C3_DIM / 128, my), 256, 0, st>>>(
        bufT, P.qkv_w + (size_t)layer * C_DIM * C3_DIM,
        P.qkv_b + (size_t)layer * C3_DIM, nullptr, nullptr, bufQKV, M, C3_DIM, C_DIM, 0);
    // attention (bf16 in/out)
    attention_kernel<<<dim3(SB, NH_DIM), 64, 0, st>>>(bufQKV, bufAttn, Ntok);
    // x += attn @ proj_w + proj_b    (fp32 stream out)
    wmma_gemm_kernel<<<dim3(C_DIM / 128, my), 256, 0, st>>>(
        bufAttn, P.proj_w + (size_t)layer * C_DIM * C_DIM,
        P.proj_b + lc, x, x, nullptr, M, C_DIM, C_DIM, 0);
    // LN2 -> bf16
    ln_bf16_kernel<<<M, 256, 0, st>>>(x, P.n2_s + lc, P.n2_b + lc, bufT, 1e-6f);
    // h1 = gelu(LN2(x) @ fc1_w + fc1_b)   (bf16 out -> fc2)
    wmma_gemm_kernel<<<dim3(HID_DIM / 128, my), 256, 0, st>>>(
        bufT, P.fc1_w + (size_t)layer * C_DIM * HID_DIM,
        P.fc1_b + (size_t)layer * HID_DIM, nullptr, nullptr, bufH1, M, HID_DIM, C_DIM, 1);
    // x += h1 @ fc2_w + fc2_b        (fp32 stream out)
    wmma_gemm_kernel<<<dim3(C_DIM / 128, my), 256, 0, st>>>(
        bufH1, P.fc2_w + (size_t)layer * HID_DIM * C_DIM,
        P.fc2_b + lc, x, x, nullptr, M, C_DIM, HID_DIM, 0);
}

static void cvt_launch(hipStream_t st, const float* src, __bf16* dst, size_t n)
{
    size_t n8 = n / 8;
    cvt_kernel<<<(unsigned)((n8 + 255) / 256), 256, 0, st>>>(src, dst, n8);
}

extern "C" void kernel_launch(void* const* d_in, const int* in_sizes, int n_in,
                              void* d_out, int out_size, void* d_ws, size_t ws_size,
                              hipStream_t stream)
{
    (void)in_sizes; (void)n_in; (void)out_size; (void)ws_size;
    // Input order: x, noise, then params flattened as a JAX pytree
    // (dict keys alphabetical, recursively).
    const float* x_in  = (const float*)d_in[0];
    const float* noise = (const float*)d_in[1];
    const float* P_embed_b   = (const float*)d_in[2];
    const float* P_embed_w   = (const float*)d_in[3];
    const float* P_head_b    = (const float*)d_in[4];
    const float* P_head_ln_b = (const float*)d_in[5];
    const float* P_head_ln_s = (const float*)d_in[6];
    const float* P_head_w    = (const float*)d_in[7];
    const float* P_snorm_b   = (const float*)d_in[8];
    const float* P_snorm_s   = (const float*)d_in[9];
    const float* P_spos      = (const float*)d_in[10];
    // ste: fc1_b fc1_w fc2_b fc2_w n1_b n1_s n2_b n2_s proj_b proj_w qkv_b qkv_w
    const float* ste_fc1_b = (const float*)d_in[11];
    const float* ste_fc1_w = (const float*)d_in[12];
    const float* ste_fc2_b = (const float*)d_in[13];
    const float* ste_fc2_w = (const float*)d_in[14];
    const float* ste_n1_b  = (const float*)d_in[15];
    const float* ste_n1_s  = (const float*)d_in[16];
    const float* ste_n2_b  = (const float*)d_in[17];
    const float* ste_n2_s  = (const float*)d_in[18];
    const float* ste_proj_b = (const float*)d_in[19];
    const float* ste_proj_w = (const float*)d_in[20];
    const float* ste_qkv_b  = (const float*)d_in[21];
    const float* ste_qkv_w  = (const float*)d_in[22];
    const float* P_tnorm_b  = (const float*)d_in[23];
    const float* P_tnorm_s  = (const float*)d_in[24];
    const float* P_tpos     = (const float*)d_in[25];
    const float* tte_fc1_b = (const float*)d_in[26];
    const float* tte_fc1_w = (const float*)d_in[27];
    const float* tte_fc2_b = (const float*)d_in[28];
    const float* tte_fc2_w = (const float*)d_in[29];
    const float* tte_n1_b  = (const float*)d_in[30];
    const float* tte_n1_s  = (const float*)d_in[31];
    const float* tte_n2_b  = (const float*)d_in[32];
    const float* tte_n2_s  = (const float*)d_in[33];
    const float* tte_proj_b = (const float*)d_in[34];
    const float* tte_proj_w = (const float*)d_in[35];
    const float* tte_qkv_b  = (const float*)d_in[36];
    const float* tte_qkv_w  = (const float*)d_in[37];

    // ---- workspace carve (all chunk sizes multiples of 8 elems, 16B align) --
    const size_t SMAX = (size_t)B_DIM * F0_DIM * N_DIM * C_DIM;  // 16.9M
    const size_t WQKV = (size_t)DEPTH_N * C_DIM * C3_DIM;
    const size_t WPRJ = (size_t)DEPTH_N * C_DIM * C_DIM;
    const size_t WFC  = (size_t)DEPTH_N * C_DIM * HID_DIM;
    char* p = (char*)d_ws;
    float* bufX = (float*)p;  p += SMAX * 4;        // stream (pre-prune)
    float* bufY = (float*)p;  p += SMAX * 4;        // stream (post-prune)
    float* bufP = (float*)p;  p += SMAX * 4;        // permuted stream
    __bf16* bufT    = (__bf16*)p; p += SMAX * 2;    // LN-out GEMM feed
    __bf16* bufQKV  = (__bf16*)p; p += 3 * SMAX * 2;
    __bf16* bufAttn = (__bf16*)p; p += SMAX * 2;
    __bf16* bufH1   = (__bf16*)p; p += 2 * SMAX * 2;
    __bf16* w_s_qkv = (__bf16*)p; p += WQKV * 2;    // bf16 weight arena
    __bf16* w_s_prj = (__bf16*)p; p += WPRJ * 2;
    __bf16* w_s_fc1 = (__bf16*)p; p += WFC * 2;
    __bf16* w_s_fc2 = (__bf16*)p; p += WFC * 2;
    __bf16* w_t_qkv = (__bf16*)p; p += WQKV * 2;
    __bf16* w_t_prj = (__bf16*)p; p += WPRJ * 2;
    __bf16* w_t_fc1 = (__bf16*)p; p += WFC * 2;
    __bf16* w_t_fc2 = (__bf16*)p; p += WFC * 2;
    float* xk      = (float*)p; p += (size_t)B_DIM * F0_DIM * C_DIM * 4;
    float* distb   = (float*)p; p += (size_t)B_DIM * F0_DIM * F0_DIM * 4;
    float* density = (float*)p; p += (size_t)B_DIM * F0_DIM * 4;
    float* dmaxb   = (float*)p; p += B_DIM * 4;
    float* scoreb  = (float*)p; p += (size_t)B_DIM * F0_DIM * 4;
    int*   idxb    = (int*)p;

    // ---- one-time (per launch) weight conversion to bf16 ----
    cvt_launch(stream, ste_qkv_w, w_s_qkv, WQKV);
    cvt_launch(stream, ste_proj_w, w_s_prj, WPRJ);
    cvt_launch(stream, ste_fc1_w, w_s_fc1, WFC);
    cvt_launch(stream, ste_fc2_w, w_s_fc2, WFC);
    cvt_launch(stream, tte_qkv_w, w_t_qkv, WQKV);
    cvt_launch(stream, tte_proj_w, w_t_prj, WPRJ);
    cvt_launch(stream, tte_fc1_w, w_t_fc1, WFC);
    cvt_launch(stream, tte_fc2_w, w_t_fc2, WFC);

    BlkP STE = { ste_n1_s, ste_n1_b, ste_qkv_b, ste_proj_b, ste_n2_s, ste_n2_b,
                 ste_fc1_b, ste_fc2_b, w_s_qkv, w_s_prj, w_s_fc1, w_s_fc2 };
    BlkP TTE = { tte_n1_s, tte_n1_b, tte_qkv_b, tte_proj_b, tte_n2_s, tte_n2_b,
                 tte_fc1_b, tte_fc2_b, w_t_qkv, w_t_prj, w_t_fc1, w_t_fc2 };

    int F = F0_DIM;
    // ---- embed + spos ----
    {
        int tokens = B_DIM * F * N_DIM;
        size_t tot = (size_t)tokens * C_DIM;
        embed_kernel<<<(unsigned)((tot + 255) / 256), 256, 0, stream>>>(
            x_in, P_embed_w, P_embed_b, P_spos, bufX, tokens);
    }

    float* cur = bufX;
    for (int i = 0; i < DEPTH_N; ++i) {
        if (i == 3) {
            // DPC-KNN frame pruning (F 243 -> 81)
            frame_mean_kernel<<<B_DIM * F, 256, 0, stream>>>(cur, xk);
            dist_kernel<<<dim3(B_DIM, F), 64, 0, stream>>>(xk, distb, F);
            density_kernel<<<B_DIM, 256, 0, stream>>>(distb, noise, density, F);
            dmax_kernel<<<B_DIM, 256, 0, stream>>>(distb, dmaxb, F);
            score_kernel<<<B_DIM, 256, 0, stream>>>(distb, density, dmaxb, scoreb, F);
            topk_kernel<<<B_DIM, 1, 0, stream>>>(scoreb, idxb, F);
            size_t gtot = (size_t)B_DIM * TOKK * N_DIM * C_DIM;
            gather_kernel<<<(unsigned)((gtot + 255) / 256), 256, 0, stream>>>(cur, idxb, bufY, F);
            cur = bufY;
            F = TOKK;
        }
        const int M = B_DIM * F * N_DIM;
        const size_t tot = (size_t)M * C_DIM;
        // spatial block on [B*F, 17, C]
        run_block(stream, cur, M, B_DIM * F, N_DIM, STE, i, bufT, bufQKV, bufAttn, bufH1);
        ln_kernel<<<M, 256, 0, stream>>>(cur, P_snorm_s, P_snorm_b, cur, 1e-6f);
        // [B,F,N,C] -> [B,N,F,C]
        permute_kernel<<<(unsigned)((tot + 255) / 256), 256, 0, stream>>>(cur, bufP, B_DIM, F, N_DIM);
        if (i == 0)
            tpos_add_kernel<<<(unsigned)((tot + 255) / 256), 256, 0, stream>>>(bufP, P_tpos, B_DIM * N_DIM, F);
        // temporal block on [B*17, F, C]
        run_block(stream, bufP, M, B_DIM * N_DIM, F, TTE, i, bufT, bufQKV, bufAttn, bufH1);
        ln_kernel<<<M, 256, 0, stream>>>(bufP, P_tnorm_s, P_tnorm_b, bufP, 1e-6f);
        // [B,N,F,C] -> [B,F,N,C]
        permute_kernel<<<(unsigned)((tot + 255) / 256), 256, 0, stream>>>(bufP, cur, B_DIM, N_DIM, F);
    }

    // ---- head: LN(eps=1e-5) -> bf16, C->3 matmul, then index as floats ----
    const int M = B_DIM * F * N_DIM;    // 8*81*17
    ln_bf16_kernel<<<M, 256, 0, stream>>>(cur, P_head_ln_s, P_head_ln_b, bufT, 1e-5f);
    float* out_f = (float*)d_out;
    head_kernel<<<(M * 3 + 255) / 256, 256, 0, stream>>>(bufT, P_head_w, P_head_b, out_f, M);
    idx_out_kernel<<<(B_DIM * TOKK + 255) / 256, 256, 0, stream>>>(idxb, out_f + (size_t)M * 3, B_DIM * TOKK);
}
